// BidirectionalLSTM_80187039416470
// MI455X (gfx1250) — matrix operand order repaired and checked by reading, not compiled, not run on hardware
//
#include <hip/hip_runtime.h>
#include <math.h>

#define WB 128
#define WT 256
#define WI 1024
#define WH 1024
#define WF 64
#define FOURH 4096
#define TWOH  2048

#if defined(__has_builtin)
#if __has_builtin(__builtin_amdgcn_global_load_async_to_lds_b128)
#define HAVE_ASYNC_LDS 1
#endif
#endif
#ifndef HAVE_ASYNC_LDS
#define HAVE_ASYNC_LDS 0
#endif

typedef __attribute__((ext_vector_type(16))) __bf16 v16bf;
typedef __attribute__((ext_vector_type(8)))  float  v8f;
typedef __attribute__((__vector_size__(16))) int    v4i;

union FragBF { v16bf v; unsigned int u[8]; };

__device__ __forceinline__ unsigned short f2bf(float f) {
  unsigned int u = __float_as_uint(f);
  u += 0x7FFFu + ((u >> 16) & 1u);           // round-to-nearest-even
  return (unsigned short)(u >> 16);
}
__device__ __forceinline__ float bf2f(unsigned short h) {
  return __uint_as_float(((unsigned int)h) << 16);
}
__device__ __forceinline__ float sigm(float x) { return 1.0f / (1.0f + __expf(-x)); }
__device__ __forceinline__ v8f vzero() {
  v8f z = {0.f,0.f,0.f,0.f,0.f,0.f,0.f,0.f}; return z;
}

#if HAVE_ASYNC_LDS
// per-lane async copy of 16B global -> LDS (ASYNCcnt-tracked, bypasses VGPRs)
__device__ __forceinline__ void async_cp16(const unsigned short* g, unsigned short* l) {
  __builtin_amdgcn_global_load_async_to_lds_b128(
      (__attribute__((address_space(1))) v4i*)(size_t)g,
      (__attribute__((address_space(3))) v4i*)(unsigned int)(size_t)l,
      0, 0);
}
#if defined(__has_builtin) && __has_builtin(__builtin_amdgcn_s_wait_asynccnt)
#define ASYNC_DRAIN() __builtin_amdgcn_s_wait_asynccnt(0)
#else
#define ASYNC_DRAIN() asm volatile("s_wait_asynccnt 0" ::: "memory")
#endif
#else
__device__ __forceinline__ void async_cp16(const unsigned short* g, unsigned short* l) {
  *(uint4*)l = *(const uint4*)g;
}
#define ASYNC_DRAIN()
#endif

// ---------------- conversion kernels ----------------

// in: [B, T, C] f32  ->  out: [T, B, C] bf16  (time-major for GEMM M tiling)
__global__ void k_transpose_bf16(const float* __restrict__ in,
                                 unsigned short* __restrict__ out, int C) {
  size_t tid = (size_t)blockIdx.x * blockDim.x + threadIdx.x;
  size_t total = (size_t)WB * WT * C;
  if (tid >= total) return;
  int c = (int)(tid % C);
  size_t bt = tid / C;
  int t = (int)(bt % WT);
  int b = (int)(bt / WT);
  out[((size_t)t * WB + b) * C + c] = f2bf(in[tid]);
}

__global__ void k_convert_bf16(const float* __restrict__ in,
                               unsigned short* __restrict__ out, size_t n) {
  size_t tid = (size_t)blockIdx.x * blockDim.x + threadIdx.x;
  if (tid < n) out[tid] = f2bf(in[tid]);
}

// zero h_n / c_n (2*B*H each) and both parity h-bf16 buffers (4*B*H)
__global__ void k_init(float* __restrict__ hn, float* __restrict__ cn,
                       unsigned short* __restrict__ hbf) {
  size_t tid = (size_t)blockIdx.x * blockDim.x + threadIdx.x;
  if (tid < (size_t)2 * WB * WH) { hn[tid] = 0.f; cn[tid] = 0.f; }
  if (tid < (size_t)4 * WB * WH) hbf[tid] = 0;
}

// ---------------- generic bf16 WMMA GEMM: D = A * Bw^T + bias ----------------
// A: [M, K] bf16 row-major; Bw: [N, K] bf16 row-major; D: [M, N] bf16.
// 128x128 tile / WG (8 waves x 16 N-cols, 8 M-tiles each), K step 32,
// double-buffered LDS with async (or reg-staged) next-tile prefetch.
__global__ void __launch_bounds__(256)
k_gemm_bf16(const unsigned short* __restrict__ A,
            const unsigned short* __restrict__ Bw0,
            const unsigned short* __restrict__ Bw1,
            const float* __restrict__ bias0, const float* __restrict__ bias1,
            unsigned short* __restrict__ D0, unsigned short* __restrict__ D1,
            int N, int K) {
  const unsigned short* Bw = blockIdx.z ? Bw1 : Bw0;
  const float* bias = blockIdx.z ? bias1 : bias0;
  unsigned short* D = blockIdx.z ? D1 : D0;

  __shared__ unsigned short As[2][128 * 32];
  __shared__ unsigned short Bs[2][128 * 32];

  const int tid = threadIdx.x;
  const int wid = tid >> 5, lane = tid & 31;
  const int ln = lane & 15, hi = lane >> 4;
  const size_t m0 = (size_t)blockIdx.y * 128;
  const size_t n0 = (size_t)blockIdx.x * 128;

  v8f acc[8];
#pragma unroll
  for (int i = 0; i < 8; i++) acc[i] = vzero();

  const int nIter = K >> 5;

  // prologue: tile 0 -> buffer 0
#pragma unroll
  for (int c = tid; c < 512; c += 256) {
    int row = c >> 2, seg = c & 3;
    async_cp16(A + (m0 + row) * (size_t)K + seg * 8, &As[0][c * 8]);
    async_cp16(Bw + (n0 + row) * (size_t)K + seg * 8, &Bs[0][c * 8]);
  }
  ASYNC_DRAIN();
  __syncthreads();

  for (int it = 0; it < nIter; ++it) {
    const int p = it & 1, pn = p ^ 1;
    const int kk2 = (it + 1) << 5;

#if HAVE_ASYNC_LDS
    if (it + 1 < nIter) {
#pragma unroll
      for (int c = tid; c < 512; c += 256) {
        int row = c >> 2, seg = c & 3;
        async_cp16(A + (m0 + row) * (size_t)K + kk2 + seg * 8, &As[pn][c * 8]);
        async_cp16(Bw + (n0 + row) * (size_t)K + kk2 + seg * 8, &Bs[pn][c * 8]);
      }
    }
#else
    uint4 ra0, ra1, rb0, rb1;
    if (it + 1 < nIter) {
      int r0 = tid >> 2, s0 = tid & 3;
      int c1 = tid + 256, r1 = c1 >> 2, s1 = c1 & 3;
      ra0 = *reinterpret_cast<const uint4*>(A + (m0 + r0) * (size_t)K + kk2 + s0 * 8);
      rb0 = *reinterpret_cast<const uint4*>(Bw + (n0 + r0) * (size_t)K + kk2 + s0 * 8);
      ra1 = *reinterpret_cast<const uint4*>(A + (m0 + r1) * (size_t)K + kk2 + s1 * 8);
      rb1 = *reinterpret_cast<const uint4*>(Bw + (n0 + r1) * (size_t)K + kk2 + s1 * 8);
    }
#endif

    // all fragment loads first, then an uninterrupted WMMA chain
    FragBF bf;
#pragma unroll
    for (int v = 0; v < 8; v++) {
      int k2 = ((v < 4) ? 2 * v : 8 + 2 * v) + hi * 8; // B: K in VGPR, N in lane
      bf.u[v] = *reinterpret_cast<const unsigned int*>(&Bs[p][(wid * 16 + ln) * 32 + k2]);
    }
    FragBF af[8];
#pragma unroll
    for (int mt = 0; mt < 8; mt++) {
      int arow = mt * 16 + ln;
#pragma unroll
      for (int v = 0; v < 8; v++) {                   // A 16x32 bf16 layout
        int klo = ((v < 4) ? 2 * v : 8 + 2 * v) + hi * 8;
        af[mt].u[v] = *reinterpret_cast<const unsigned int*>(&As[p][arow * 32 + klo]);
      }
    }
#pragma unroll
    for (int mt = 0; mt < 8; mt++)
      acc[mt] = __builtin_amdgcn_wmma_f32_16x16x32_bf16(
          false, af[mt].v, false, bf.v, (short)0, acc[mt], false, false);

#if !HAVE_ASYNC_LDS
    if (it + 1 < nIter) {
      ((uint4*)As[pn])[tid] = ra0;       ((uint4*)Bs[pn])[tid] = rb0;
      ((uint4*)As[pn])[tid + 256] = ra1; ((uint4*)Bs[pn])[tid + 256] = rb1;
    }
#endif
    ASYNC_DRAIN();
    __syncthreads();
  }

  size_t n = n0 + wid * 16 + ln;
  float bv = bias ? bias[n] : 0.0f;
#pragma unroll
  for (int mt = 0; mt < 8; mt++) {
#pragma unroll
    for (int r = 0; r < 8; r++) {                     // C/D: m = r + 8*hi
      size_t m = m0 + mt * 16 + r + hi * 8;
      D[m * (size_t)N + n] = f2bf(acc[mt][r] + bv);
    }
  }
}

// fg: [2, T*B, 2H] bf16  ->  ld: [2, T*B, H] f32 ; ld = sigmoid(lo)*tanh(hi)
__global__ void k_fieldgate_act(const unsigned short* __restrict__ fg,
                                float* __restrict__ ld) {
  size_t tid = (size_t)blockIdx.x * blockDim.x + threadIdx.x;
  size_t total = (size_t)2 * WT * WB * WH;
  if (tid >= total) return;
  int j = (int)(tid % WH);
  size_t rowd = tid / WH;
  const unsigned short* f = fg + rowd * TWOH;
  ld[tid] = sigm(bf2f(f[j])) * tanhf(bf2f(f[WH + j]));
}

// ---------------- fused recurrent step ----------------
// grid: (H/16, 1, 2 dirs). Each WG owns 16 hidden cols c0..c0+15, computes the
// four matching gate blocks of h@W_hh^T (rows g*1024+c0+j of W_hh) via WMMA,
// stages to LDS, then does the full LSTM cell update for (all b, those cols).
__global__ void __launch_bounds__(256)
k_lstm_step(int step,
            const unsigned short* __restrict__ hbf_in,   // [2,B,H] bf16
            unsigned short* __restrict__ hbf_out,        // [2,B,H] bf16
            const unsigned short* __restrict__ WhhB,     // [2,4H,H] bf16
            const unsigned short* __restrict__ preG,     // [2,T*B,4H] bf16
            const float* __restrict__ ldt,               // [2,T*B,H] f32
            float* __restrict__ outputs,                 // [T*B, 2H]
            float* __restrict__ h_n,                     // [2,B,H]
            float* __restrict__ c_n) {                   // [2,B,H]
  const int dir = blockIdx.z;
  const int t = dir ? (WT - 1 - step) : step;
  const int c0 = blockIdx.x * 16;

  __shared__ unsigned short As[2][128 * 32];   // h tile   (2 x 8 KB)
  __shared__ unsigned short Bs[2][64 * 32];    // W tile   (2 x 4 KB)
  __shared__ float gl[4 * 128 * 16];           // gate acc (32 KB)

  const int tid = threadIdx.x;
  const int wid = tid >> 5, lane = tid & 31;
  const int ln = lane & 15, hi = lane >> 4;
  const int g = wid >> 1;                   // gate block 0..3 (i,f,g,o)
  const int mh = (wid & 1) * 64;            // batch half

  const unsigned short* A = hbf_in + (size_t)dir * WB * WH;
  const unsigned short* W = WhhB + (size_t)dir * FOURH * WH;

  // B-row for this thread's W-tile chunk (64 rows x 4 segs = 256 chunks)
  const int blr = tid >> 2, bseg = tid & 3;
  const int bgg = blr >> 4, bj = blr & 15;
  const size_t bwrow = (size_t)((bgg << 10) + c0 + bj) * WH + bseg * 8;

  v8f acc[4];
#pragma unroll
  for (int i = 0; i < 4; i++) acc[i] = vzero();

  const int nIter = WH >> 5;

  // prologue: tile 0 -> buffer 0
#pragma unroll
  for (int c = tid; c < 512; c += 256) {
    int row = c >> 2, seg = c & 3;
    async_cp16(A + (size_t)row * WH + seg * 8, &As[0][c * 8]);
  }
  async_cp16(W + bwrow, &Bs[0][tid * 8]);
  ASYNC_DRAIN();
  __syncthreads();

  for (int it = 0; it < nIter; ++it) {
    const int p = it & 1, pn = p ^ 1;
    const int kk2 = (it + 1) << 5;

#if HAVE_ASYNC_LDS
    if (it + 1 < nIter) {
#pragma unroll
      for (int c = tid; c < 512; c += 256) {
        int row = c >> 2, seg = c & 3;
        async_cp16(A + (size_t)row * WH + kk2 + seg * 8, &As[pn][c * 8]);
      }
      async_cp16(W + bwrow + kk2, &Bs[pn][tid * 8]);
    }
#else
    uint4 ra0, ra1, rb0;
    if (it + 1 < nIter) {
      int r0 = tid >> 2, s0 = tid & 3;
      int c1 = tid + 256, r1 = c1 >> 2, s1 = c1 & 3;
      ra0 = *reinterpret_cast<const uint4*>(A + (size_t)r0 * WH + kk2 + s0 * 8);
      ra1 = *reinterpret_cast<const uint4*>(A + (size_t)r1 * WH + kk2 + s1 * 8);
      rb0 = *reinterpret_cast<const uint4*>(W + bwrow + kk2);
    }
#endif

    FragBF bf;
#pragma unroll
    for (int v = 0; v < 8; v++) {
      int k2 = ((v < 4) ? 2 * v : 8 + 2 * v) + hi * 8;
      bf.u[v] = *reinterpret_cast<const unsigned int*>(&Bs[p][(g * 16 + ln) * 32 + k2]);
    }
    FragBF af[4];
#pragma unroll
    for (int mt = 0; mt < 4; mt++) {
      int arow = mh + mt * 16 + ln;
#pragma unroll
      for (int v = 0; v < 8; v++) {
        int klo = ((v < 4) ? 2 * v : 8 + 2 * v) + hi * 8;
        af[mt].u[v] = *reinterpret_cast<const unsigned int*>(&As[p][arow * 32 + klo]);
      }
    }
#pragma unroll
    for (int mt = 0; mt < 4; mt++)
      acc[mt] = __builtin_amdgcn_wmma_f32_16x16x32_bf16(
          false, af[mt].v, false, bf.v, (short)0, acc[mt], false, false);

#if !HAVE_ASYNC_LDS
    if (it + 1 < nIter) {
      ((uint4*)As[pn])[tid] = ra0;
      ((uint4*)As[pn])[tid + 256] = ra1;
      ((uint4*)Bs[pn])[tid] = rb0;
    }
#endif
    ASYNC_DRAIN();
    __syncthreads();
  }

  // stage gate partials: gl[g][m][n]
#pragma unroll
  for (int mt = 0; mt < 4; mt++)
#pragma unroll
    for (int r = 0; r < 8; r++) {
      int m = mh + mt * 16 + r + hi * 8;
      gl[g * 2048 + m * 16 + ln] = acc[mt][r];
    }
  __syncthreads();

  // cell update: 128 b x 16 cols = 2048 elems, 8 per thread
#pragma unroll
  for (int q = 0; q < 8; q++) {
    int e = tid + 256 * q;
    int b = e >> 4, jj = e & 15;
    int j = c0 + jj;
    size_t row = (size_t)t * WB + b;
    const unsigned short* pg = preG + ((size_t)dir * WT * WB + row) * FOURH;
    float gi = bf2f(pg[j])           + gl[0 * 2048 + b * 16 + jj];
    float gf = bf2f(pg[WH + j])      + gl[1 * 2048 + b * 16 + jj];
    float gg = bf2f(pg[2 * WH + j])  + gl[2 * 2048 + b * 16 + jj];
    float go = bf2f(pg[3 * WH + j])  + gl[3 * 2048 + b * 16 + jj];
    float ig = sigm(gi), fgt = sigm(gf), gv = tanhf(gg), og = sigm(go);
    float lv = ldt[((size_t)dir * WT * WB + row) * WH + j];
    size_t sidx = (size_t)dir * WB * WH + (size_t)b * WH + j;
    float cc = fgt * c_n[sidx] + ig * gv + lv;
    float hh = og * tanhf(cc);
    c_n[sidx] = cc;
    h_n[sidx] = hh;
    hbf_out[sidx] = f2bf(hh);
    outputs[row * (size_t)TWOH + (size_t)dir * WH + j] = hh;
  }
}

// ---------------- host launch ----------------
extern "C" void kernel_launch(void* const* d_in, const int* in_sizes, int n_in,
                              void* d_out, int out_size, void* d_ws, size_t ws_size,
                              hipStream_t stream) {
  const float* seq  = (const float*)d_in[0];
  const float* fpos = (const float*)d_in[1];
  const float* Wihf = (const float*)d_in[2];
  const float* Whhf = (const float*)d_in[3];
  const float* bf_  = (const float*)d_in[4];
  const float* Wfpf = (const float*)d_in[5];
  const float* bfpf = (const float*)d_in[6];
  const float* Wihb = (const float*)d_in[7];
  const float* Whhb = (const float*)d_in[8];
  const float* bb_  = (const float*)d_in[9];
  const float* Wfpb = (const float*)d_in[10];
  const float* bfpb = (const float*)d_in[11];

  char* ws = (char*)d_ws;
  size_t off = 0;
  auto alloc = [&](size_t bytes) -> char* {
    char* p = ws + off;
    off += (bytes + 255) & ~(size_t)255;
    return p;
  };
  unsigned short* Xb   = (unsigned short*)alloc((size_t)WT * WB * WI * 2);
  unsigned short* FPb  = (unsigned short*)alloc((size_t)WT * WB * WF * 2);
  unsigned short* WihB = (unsigned short*)alloc((size_t)2 * FOURH * WI * 2);
  unsigned short* WhhB = (unsigned short*)alloc((size_t)2 * FOURH * WH * 2);
  unsigned short* WfpB = (unsigned short*)alloc((size_t)2 * TWOH * WF * 2);
  unsigned short* preG = (unsigned short*)alloc((size_t)2 * WT * WB * FOURH * 2);
  unsigned short* fgT  = (unsigned short*)alloc((size_t)2 * WT * WB * TWOH * 2);
  float*          ldt  = (float*)alloc((size_t)2 * WT * WB * WH * 4);
  unsigned short* hbf  = (unsigned short*)alloc((size_t)4 * WB * WH * 2); // ping-pong x 2 dirs

  float* outputs = (float*)d_out;
  float* h_n = outputs + (size_t)WT * WB * TWOH;
  float* c_n = h_n + (size_t)2 * WB * WH;

  // 1) convert / transpose everything to bf16 once
  k_transpose_bf16<<<(WB * WT * WI) / 256, 256, 0, stream>>>(seq, Xb, WI);
  k_transpose_bf16<<<(WB * WT * WF) / 256, 256, 0, stream>>>(fpos, FPb, WF);
  k_convert_bf16<<<(FOURH * WI) / 256, 256, 0, stream>>>(Wihf, WihB, (size_t)FOURH * WI);
  k_convert_bf16<<<(FOURH * WI) / 256, 256, 0, stream>>>(Wihb, WihB + (size_t)FOURH * WI, (size_t)FOURH * WI);
  k_convert_bf16<<<(FOURH * WH) / 256, 256, 0, stream>>>(Whhf, WhhB, (size_t)FOURH * WH);
  k_convert_bf16<<<(FOURH * WH) / 256, 256, 0, stream>>>(Whhb, WhhB + (size_t)FOURH * WH, (size_t)FOURH * WH);
  k_convert_bf16<<<(TWOH * WF) / 256, 256, 0, stream>>>(Wfpf, WfpB, (size_t)TWOH * WF);
  k_convert_bf16<<<(TWOH * WF) / 256, 256, 0, stream>>>(Wfpb, WfpB + (size_t)TWOH * WF, (size_t)TWOH * WF);
  k_init<<<(4 * WB * WH) / 256, 256, 0, stream>>>(h_n, c_n, hbf);

  // 2) time-parallel WMMA GEMMs: pre-gates (M=32768,N=4096,K=1024)
  k_gemm_bf16<<<dim3(FOURH / 128, (WT * WB) / 128, 2), 256, 0, stream>>>(
      Xb, WihB, WihB + (size_t)FOURH * WI, bf_, bb_,
      preG, preG + (size_t)WT * WB * FOURH, FOURH, WI);

  // field gates (M=32768,N=2048,K=64), then activation -> ld term
  k_gemm_bf16<<<dim3(TWOH / 128, (WT * WB) / 128, 2), 256, 0, stream>>>(
      FPb, WfpB, WfpB + (size_t)TWOH * WF, bfpf, bfpb,
      fgT, fgT + (size_t)WT * WB * TWOH, TWOH, WF);
  k_fieldgate_act<<<((size_t)2 * WT * WB * WH) / 256, 256, 0, stream>>>(fgT, ldt);

  // 3) sequential recurrence, both directions fused per launch
  for (int s = 0; s < WT; s++) {
    const unsigned short* hin = hbf + (size_t)(s & 1) * 2 * WB * WH;
    unsigned short* hout = hbf + (size_t)((s & 1) ^ 1) * 2 * WB * WH;
    k_lstm_step<<<dim3(WH / 16, 1, 2), 256, 0, stream>>>(
        s, hin, hout, WhhB, preG, ldt, outputs, h_n, c_n);
  }
}
